// SparseMoEBlock_14276471291957
// MI455X (gfx1250) — matrix operand order repaired and checked by reading, not compiled, not run on hardware
//
#include <hip/hip_runtime.h>

// ---------------------------------------------------------------------------
// Sparse MoE block (top-2 of 8 experts, SwiGLU FFN) for MI455X (gfx1250).
// router -> per-expert token gather -> two fp16-WMMA GEMMs per expert with
// fp32 accumulation. fp32->fp16 conversion happens in-register while staging
// tiles into double-buffered LDS; fragments are preloaded so WMMAs issue
// back-to-back instead of stalling on per-op s_wait_dscnt.
// Deterministic: no atomics, fixed traversal order, kernels serialized.
// ---------------------------------------------------------------------------

#define T_TOK 8192     // B*S tokens
#define DDIM  2048     // model dim
#define HDIM  1408     // ffn hidden dim
#define EXP   8        // experts

typedef __attribute__((ext_vector_type(16))) _Float16 v16h;
typedef __attribute__((ext_vector_type(8)))  float    v8f;

union Frag { v16h h; uint4 u[2]; };

__device__ __forceinline__ unsigned pk2(float a, float b) {
  union { _Float16 h[2]; unsigned u; } v;
  v.h[0] = (_Float16)a; v.h[1] = (_Float16)b;
  return v.u;
}

// Convert 8 contiguous fp32 -> 8 fp16 and store as one 16B LDS chunk.
__device__ __forceinline__ void cvt8(const float* __restrict__ s,
                                     unsigned short* d) {
  float4 a = *(const float4*)s;
  float4 b = *(const float4*)(s + 4);
  uint4 q;
  q.x = pk2(a.x, a.y); q.y = pk2(a.z, a.w);
  q.z = pk2(b.x, b.y); q.w = pk2(b.z, b.w);
  *(uint4*)d = q;
}

// ---------------------------------------------------------------------------
// Kernel 0: zero the output accumulator.
// ---------------------------------------------------------------------------
__global__ __launch_bounds__(256) void zero_f4(float4* __restrict__ p, int n4) {
  int i = blockIdx.x * 256 + threadIdx.x;
  if (i < n4) p[i] = make_float4(0.f, 0.f, 0.f, 0.f);
}

// ---------------------------------------------------------------------------
// Kernel 1: router. One wave32 per token; router weights (8x2048 fp32 = 64KB)
// cached in LDS. fp32 logits -> softmax -> top-2 -> renormalize.
// ---------------------------------------------------------------------------
__global__ __launch_bounds__(256) void router_kernel(
    const float* __restrict__ x, const float* __restrict__ rw,
    int* __restrict__ idx2, float* __restrict__ w2) {
  __shared__ float srw[EXP * DDIM];  // 64 KB (LDS is 320 KB/WGP on CDNA5)
  const int tid = threadIdx.x;
  for (int j = tid; j < EXP * DDIM; j += 256) srw[j] = rw[j];
  __syncthreads();

  const int wave = tid >> 5, lane = tid & 31;
  const int t = blockIdx.x * 8 + wave;
  const float* xr = x + (size_t)t * DDIM;

  float acc[EXP];
#pragma unroll
  for (int e = 0; e < EXP; ++e) acc[e] = 0.f;

  for (int d = lane; d < DDIM; d += 32) {
    float xv = xr[d];
#pragma unroll
    for (int e = 0; e < EXP; ++e) acc[e] += xv * srw[e * DDIM + d];
  }
#pragma unroll
  for (int off = 16; off > 0; off >>= 1) {
#pragma unroll
    for (int e = 0; e < EXP; ++e) acc[e] += __shfl_xor(acc[e], off, 32);
  }

  if (lane == 0) {
    float m = acc[0];
#pragma unroll
    for (int e = 1; e < EXP; ++e) m = fmaxf(m, acc[e]);
    float ex[EXP];
#pragma unroll
    for (int e = 0; e < EXP; ++e) ex[e] = __expf(acc[e] - m);
    // top-2 (lowest index wins ties, matching lax.top_k)
    int i0 = 0; float b0 = ex[0];
#pragma unroll
    for (int e = 1; e < EXP; ++e) if (ex[e] > b0) { b0 = ex[e]; i0 = e; }
    int i1 = (i0 == 0) ? 1 : 0; float b1 = ex[i1];
#pragma unroll
    for (int e = 0; e < EXP; ++e)
      if (e != i0 && e != ((i0 == 0) ? 1 : 0) && ex[e] > b1) { b1 = ex[e]; i1 = e; }
    const float inv = 1.f / (b0 + b1);   // softmax denom cancels in renorm
    idx2[2 * t]     = i0;
    idx2[2 * t + 1] = i1;
    w2[2 * t]       = b0 * inv;
    w2[2 * t + 1]   = b1 * inv;
  }
}

// ---------------------------------------------------------------------------
// Kernel 2: per-expert token list compaction (ascending token order ->
// deterministic). One wave per expert; ballot + popc prefix.
// ---------------------------------------------------------------------------
__global__ __launch_bounds__(32) void build_lists(
    const int* __restrict__ idx2, const float* __restrict__ w2,
    int* __restrict__ lists, float* __restrict__ wl, int* __restrict__ counts) {
  const int e = blockIdx.x;
  const int lane = threadIdx.x;
  int cnt = 0;
  for (int base = 0; base < T_TOK; base += 32) {
    const int t = base + lane;
    const int i0 = idx2[2 * t], i1 = idx2[2 * t + 1];
    const bool sel = (i0 == e) || (i1 == e);
    const float w = (i0 == e) ? w2[2 * t] : w2[2 * t + 1];
#if __has_builtin(__builtin_amdgcn_ballot_w32)
    unsigned mask = __builtin_amdgcn_ballot_w32(sel);
#else
    unsigned mask = (unsigned)__ballot(sel);
#endif
    const int pos = __popc(mask & ((1u << lane) - 1u));
    if (sel) {
      lists[e * T_TOK + cnt + pos] = t;
      wl[e * T_TOK + cnt + pos] = w;
    }
    cnt += __popc(mask);
  }
  if (lane == 0) counts[e] = cnt;
}

// ---------------------------------------------------------------------------
// Kernel 3: gathered GEMM1 + SwiGLU.
//   hidden[r, :H] = silu(x[tok] @ Wg^T) * (x[tok] @ Wu^T)
// Block = 8 waves = 128 gathered rows x (64 gate + matched 64 up) cols.
// Double-buffered LDS; next K-step staged between frag loads and WMMA chain.
// ---------------------------------------------------------------------------
__global__ __launch_bounds__(256) void gemm1_swiglu(
    const float* __restrict__ xf, const float* __restrict__ gup,
    const int* __restrict__ lists, const int* __restrict__ counts,
    unsigned short* __restrict__ hidden, int e) {
  __shared__ int stok[128];
  __shared__ __align__(16) unsigned short sA[2][128 * 32];   // 2 x 8 KB
  __shared__ __align__(16) unsigned short sBg[2][64 * 32];   // 2 x 4 KB
  __shared__ __align__(16) unsigned short sBu[2][64 * 32];   // 2 x 4 KB

  const int cnt = counts[e];
  const int rowBase = blockIdx.x * 128;
  if (rowBase >= cnt) return;                 // uniform exit: EXEC stays full
  const int colBase = blockIdx.y * 64;
  const int tid = threadIdx.x;
  const float* gupE = gup + (size_t)e * (2 * HDIM) * DDIM;
  const int* listE = lists + e * T_TOK;

  if (tid < 128) {
    const int r = rowBase + tid;
    stok[tid] = listE[r < cnt ? r : (cnt - 1)];
  }
  __syncthreads();

  const int lane = tid & 31, wave = tid >> 5;
  v8f accG[4], accU[4];
#pragma unroll
  for (int nt = 0; nt < 4; ++nt) {
#pragma unroll
    for (int i = 0; i < 8; ++i) { accG[nt][i] = 0.f; accU[nt][i] = 0.f; }
  }

  // Staging maps (all 256 threads participate each K-step).
  const int sr = tid >> 1;                 // 0..127 : A row in block
  const int sh = (tid & 1) * 16;           // 0 or 16 : K half
  const size_t aSrc = (size_t)stok[sr] * DDIM;
  const int br = tid >> 1;                 // 0..127 : 0-63 gate rows, 64-127 up
  const size_t bSrc = (br < 64)
      ? (size_t)(colBase + br) * DDIM
      : (size_t)(HDIM + colBase + (br - 64)) * DDIM;
  const int bOff = (br & 63) * 32 + sh;
  const bool isGate = (br < 64);

  // Fragment addresses.
  const int arow = wave * 16 + (lane & 15);
  const int ak = (lane < 16) ? 0 : 8;      // A: K{0..7,16..23} / K{8..15,24..31}
  const int bc = lane & 15;
  const int bk = (lane < 16) ? 0 : 8;      // B: K{0..7,16..23} / K{8..15,24..31}

  // Prologue: stage K-step 0 into buffer 0.
  {
    cvt8(xf + aSrc + sh,     &sA[0][sr * 32 + sh]);
    cvt8(xf + aSrc + sh + 8, &sA[0][sr * 32 + sh + 8]);
    unsigned short* bD = (isGate ? sBg[0] : sBu[0]) + bOff;
    cvt8(gupE + bSrc + sh,     bD);
    cvt8(gupE + bSrc + sh + 8, bD + 8);
  }

  for (int k0 = 0; k0 < DDIM; k0 += 32) {
    const int buf = (k0 >> 5) & 1;
    __syncthreads();   // staging of `buf` complete; prior reads of buf^1 done

    // Preload ALL fragments for this K-step (ds_loads pipeline together).
    Frag a, bg[4], bu[4];
    const unsigned short* pA = &sA[buf][arow * 32 + ak];
    a.u[0] = *(const uint4*)pA;
    a.u[1] = *(const uint4*)(pA + 16);
#pragma unroll
    for (int nt = 0; nt < 4; ++nt) {
      const int o = (nt * 16 + bc) * 32 + bk;
      bg[nt].u[0] = *(const uint4*)&sBg[buf][o];
      bg[nt].u[1] = *(const uint4*)&sBg[buf][o + 16];
      bu[nt].u[0] = *(const uint4*)&sBu[buf][o];
      bu[nt].u[1] = *(const uint4*)&sBu[buf][o + 16];
    }

    // Stage next K-step into the alternate buffer (overlaps with WMMAs).
    if (k0 + 32 < DDIM) {
      const int nb = buf ^ 1;
      const int kn = k0 + 32;
      cvt8(xf + aSrc + kn + sh,     &sA[nb][sr * 32 + sh]);
      cvt8(xf + aSrc + kn + sh + 8, &sA[nb][sr * 32 + sh + 8]);
      unsigned short* bD = (isGate ? sBg[nb] : sBu[nb]) + bOff;
      cvt8(gupE + bSrc + kn + sh,     bD);
      cvt8(gupE + bSrc + kn + sh + 8, bD + 8);
    }

    // Back-to-back WMMA chain (no per-op LDS waits).
#pragma unroll
    for (int nt = 0; nt < 4; ++nt) {
      accG[nt] = __builtin_amdgcn_wmma_f32_16x16x32_f16(
          false, a.h, false, bg[nt].h, (short)0, accG[nt], false, false);
      accU[nt] = __builtin_amdgcn_wmma_f32_16x16x32_f16(
          false, a.h, false, bu[nt].h, (short)0, accU[nt], false, false);
    }
  }

  // SwiGLU on accumulators (gate/up tiles share the lane<->element map).
  const int mOff = (lane < 16) ? 0 : 8;
  const int nCol = lane & 15;
#pragma unroll
  for (int nt = 0; nt < 4; ++nt) {
#pragma unroll
    for (int i = 0; i < 8; ++i) {
      const int gr = rowBase + wave * 16 + mOff + i;
      if (gr < cnt) {
        const float g = accG[nt][i];
        const float u = accU[nt][i];
        const float hv = (g / (1.f + __expf(-g))) * u;
        union { _Float16 h; unsigned short us; } c;
        c.h = (_Float16)hv;
        hidden[(size_t)gr * HDIM + colBase + nt * 16 + nCol] = c.us;
      }
    }
  }
}

// ---------------------------------------------------------------------------
// Kernel 4: GEMM2 (down proj) + weighted scatter-accumulate into out.
//   out[tok, :] += w_tok * (hidden[r, :H] @ Wd^T)
// Block = 128 rows x 128 cols (8 WMMA tiles per wave). Double-buffered B.
// ---------------------------------------------------------------------------
__global__ __launch_bounds__(256) void gemm2_down(
    const unsigned short* __restrict__ hidden, const float* __restrict__ dp,
    const int* __restrict__ lists, const float* __restrict__ wl,
    const int* __restrict__ counts, float* __restrict__ out, int e) {
  __shared__ int stok[128];
  __shared__ float swt[128];
  __shared__ __align__(16) unsigned short sB[2][128 * 32];   // 2 x 8 KB

  const int cnt = counts[e];
  const int rowBase = blockIdx.x * 128;
  if (rowBase >= cnt) return;
  const int colBase = blockIdx.y * 128;
  const int tid = threadIdx.x;
  const float* dpE = dp + (size_t)e * DDIM * HDIM;

  if (tid < 128) {
    const int r = rowBase + tid;
    const int rc = (r < cnt) ? r : (cnt - 1);
    stok[tid] = lists[e * T_TOK + rc];
    swt[tid] = (r < cnt) ? wl[e * T_TOK + rc] : 0.f;
  }
  __syncthreads();

  const int lane = tid & 31, wave = tid >> 5;
  v8f acc[8];
#pragma unroll
  for (int nt = 0; nt < 8; ++nt) {
#pragma unroll
    for (int i = 0; i < 8; ++i) acc[nt][i] = 0.f;
  }

  int arowG = rowBase + wave * 16 + (lane & 15);
  if (arowG >= cnt) arowG = cnt - 1;         // safe read; store is guarded
  const unsigned short* aRow = hidden + (size_t)arowG * HDIM;
  const int ak = (lane < 16) ? 0 : 8;

  // B staging: 128 rows (cols of D) x 32 K per step.
  const int br = tid >> 1;                   // 0..127
  const int sh = (tid & 1) * 16;             // K half
  const size_t bSrc = (size_t)(colBase + br) * HDIM;
  const int bOff = br * 32 + sh;

  const int bc = lane & 15;
  const int bk = (lane < 16) ? 0 : 8;

  // Prologue: stage K-step 0.
  cvt8(dpE + bSrc + sh,     &sB[0][bOff]);
  cvt8(dpE + bSrc + sh + 8, &sB[0][bOff + 8]);

  for (int k0 = 0; k0 < HDIM; k0 += 32) {
    const int buf = (k0 >> 5) & 1;
    __syncthreads();

    // Preload all fragments: A direct from global (compacted fp16 rows),
    // B from LDS.
    Frag a, b[8];
    a.u[0] = *(const uint4*)(aRow + k0 + ak);
    a.u[1] = *(const uint4*)(aRow + k0 + ak + 16);
#pragma unroll
    for (int nt = 0; nt < 8; ++nt) {
      const int o = (nt * 16 + bc) * 32 + bk;
      b[nt].u[0] = *(const uint4*)&sB[buf][o];
      b[nt].u[1] = *(const uint4*)&sB[buf][o + 16];
    }

    // Stage next K-step into alternate buffer.
    if (k0 + 32 < HDIM) {
      const int nb = buf ^ 1;
      const int kn = k0 + 32;
      cvt8(dpE + bSrc + kn + sh,     &sB[nb][bOff]);
      cvt8(dpE + bSrc + kn + sh + 8, &sB[nb][bOff + 8]);
    }

#pragma unroll
    for (int nt = 0; nt < 8; ++nt) {
      acc[nt] = __builtin_amdgcn_wmma_f32_16x16x32_f16(
          false, a.h, false, b[nt].h, (short)0, acc[nt], false, false);
    }
  }

  // Weighted accumulate into out (kernels serialize on the stream; each token
  // appears in exactly two expert lists -> two commutative fp32 adds).
  const int mOff = (lane < 16) ? 0 : 8;
  const int nCol = lane & 15;
#pragma unroll
  for (int i = 0; i < 8; ++i) {
    const int lrow = wave * 16 + mOff + i;
    const int gr = rowBase + lrow;
    if (gr < cnt) {
      const int tok = stok[lrow];
      const float w = swt[lrow];
#pragma unroll
      for (int nt = 0; nt < 8; ++nt) {
        const size_t o = (size_t)tok * DDIM + colBase + nt * 16 + nCol;
        out[o] += w * acc[nt][i];
      }
    }
  }
}

// ---------------------------------------------------------------------------
// Host-side launcher.
// ---------------------------------------------------------------------------
extern "C" void kernel_launch(void* const* d_in, const int* in_sizes, int n_in,
                              void* d_out, int out_size, void* d_ws,
                              size_t ws_size, hipStream_t stream) {
  const float* x   = (const float*)d_in[0];   // [T, D]
  const float* gup = (const float*)d_in[1];   // [E, 2H, D]
  const float* dp  = (const float*)d_in[2];   // [E, D, H]
  const float* rw  = (const float*)d_in[3];   // [E, D]
  float* out = (float*)d_out;                 // [T, D]

  // Workspace carve-up (~23.7 MB total).
  char* ws = (char*)d_ws;
  size_t o = 0;
  unsigned short* hidden = (unsigned short*)(ws + o); o += (size_t)T_TOK * HDIM * 2;
  int*   lists  = (int*)(ws + o);   o += (size_t)EXP * T_TOK * 4;
  float* wl     = (float*)(ws + o); o += (size_t)EXP * T_TOK * 4;
  int*   counts = (int*)(ws + o);   o += 256;
  int*   idx2   = (int*)(ws + o);   o += (size_t)T_TOK * 2 * 4;
  float* w2     = (float*)(ws + o); o += (size_t)T_TOK * 2 * 4;

  const int n4 = (T_TOK * DDIM) / 4;
  zero_f4<<<(n4 + 255) / 256, 256, 0, stream>>>((float4*)out, n4);

  router_kernel<<<T_TOK / 8, 256, 0, stream>>>(x, rw, idx2, w2);
  build_lists<<<EXP, 32, 0, stream>>>(idx2, w2, lists, wl, counts);

  const dim3 g1(T_TOK / 128, HDIM / 64);    // (64, 22), worst case rows
  const dim3 g2(T_TOK / 128, DDIM / 128);   // (64, 16)
  for (int e = 0; e < EXP; ++e) {
    gemm1_swiglu<<<g1, 256, 0, stream>>>(x, gup, lists, counts, hidden, e);
    gemm2_down<<<g2, 256, 0, stream>>>(hidden, dp, lists, wl, counts, out, e);
  }
}
